// RSSM_17308718203731
// MI455X (gfx1250) — compile-verified, matrix operand order checked by reading, not run on hardware
//
#include <hip/hip_runtime.h>
#include <stdint.h>

// RSSM on gfx1250: bf16 WMMA GEMMs with fp32 accumulate, fused GRU + heads,
// double-buffered async global->LDS staging pipelined against WMMA.
// B=256, T=64, S=32, D=1024, H=1024, L=1536, A=32.

#define Bx 256
#define Tx 64
#define Sx 32
#define Dx 1024
#define Hx 1024
#define Lx 1536
#define Ax 32
#define FEATD (Sx + Dx)   // 1056

// Flip to 0 if the integrated assembler rejects the async-to-LDS mnemonic.
#define USE_ASYNC_LDS 1

typedef __attribute__((ext_vector_type(16))) __bf16 v16bf;
typedef __attribute__((ext_vector_type(8)))  float  v8f;

union FragU { v16bf v; uint4 q[2]; };

#define WMMA_BF16(a, b, c) \
  __builtin_amdgcn_wmma_f32_16x16x32_bf16(false, (a), false, (b), (short)0, (c), false, false)

// A-fragment (16x32, bf16) per CDNA5 layout:
// lanes 0-15: row M=lane, elems 0-7 = K 0..7, elems 8-15 = K 16..23
// lanes 16-31: row M=lane-16, elems 0-7 = K 8..15, elems 8-15 = K 24..31
__device__ __forceinline__ v16bf frag_lds(const __bf16* s, int lane) {
  int r  = lane & 15;
  int kb = (lane >> 4) << 3;
  FragU u;
  u.q[0] = *(const uint4*)(s + r * 32 + kb);
  u.q[1] = *(const uint4*)(s + r * 32 + kb + 16);
  return u.v;
}

// B-fragment (32x16): W stored row-major [N,K]; B[k][n] = W[n][k], so a
// column of B is a contiguous row of W. Same per-lane byte pattern as A.
__device__ __forceinline__ v16bf frag_glb(const __bf16* __restrict__ W, int ldw,
                                          int n0, int k0, int lane) {
  int n  = n0 + (lane & 15);
  int kb = k0 + ((lane >> 4) << 3);
  const __bf16* p = W + (size_t)n * ldw + kb;
  FragU u;
  u.q[0] = *(const uint4*)(p);
  u.q[1] = *(const uint4*)(p + 16);
  return u.v;
}

// Stage a 16x32 bf16 tile of A (row stride lda) into LDS. 64 lanes x 16B.
// Async variant: GLOBAL_LOAD_ASYNC_TO_LDS_B128 (GV mode), tracked by ASYNCcnt,
// overlaps with WMMA; completion enforced in wait_stage().
__device__ __forceinline__ void stage_tile(__bf16* dst, const __bf16* __restrict__ A,
                                           int lda, int m0, int k0, int tid) {
  if (tid < 64) {
    int r = tid >> 2, c = (tid & 3) << 3;
    const __bf16* g = A + (size_t)(m0 + r) * lda + k0 + c;
#if USE_ASYNC_LDS
    uint32_t lds = (uint32_t)(uintptr_t)(dst + r * 32 + c);
    asm volatile("global_load_async_to_lds_b128 %0, %1, off"
                 :: "v"(lds), "v"(g) : "memory");
#else
    *(uint4*)(dst + r * 32 + c) = *(const uint4*)g;
#endif
  }
}

__device__ __forceinline__ void wait_stage() {
#if USE_ASYNC_LDS
  asm volatile("s_wait_asynccnt 0" ::: "memory");
#endif
  __syncthreads();
}

__device__ __forceinline__ float sigmoid_f(float x) { return 1.0f / (1.0f + __expf(-x)); }
__device__ __forceinline__ float elu_f(float x)     { return x > 0.0f ? x : __expf(x) - 1.0f; }
__device__ __forceinline__ float softplus_f(float x){ return fmaxf(x, 0.0f) + log1pf(__expf(-fabsf(x))); }

// ---------------- C = elu(A[256xK]bf16 @ W[N,K]^T + bias) -> bf16 ----------------
// block: 256 threads (8 waves). Wave owns a 16x64 strip: 4 accumulators share
// one A-fragment -> 4 WMMAs per K-chunk per wave, 1 barrier per chunk.
// grid: (N/512, 16).
__global__ void gemm_bias_elu_kernel(const __bf16* __restrict__ A, int K,
                                     const __bf16* __restrict__ W,
                                     const float* __restrict__ bias,
                                     __bf16* __restrict__ C, int ldc) {
  __shared__ __align__(16) __bf16 sA[2][16 * 32];
  const int tid  = threadIdx.x;
  const int lane = tid & 31;
  const int wave = tid >> 5;
  const int m0   = blockIdx.y * 16;
  const int n0   = (blockIdx.x * 8 + wave) * 64;

  v8f acc[4] = {};
  stage_tile(sA[0], A, K, m0, 0, tid);
  wait_stage();

  const int nIter = K >> 5;
  for (int i = 0; i < nIter; ++i) {
    const int k0  = i << 5;
    const int buf = i & 1;
    if (i + 1 < nIter) {
      stage_tile(sA[buf ^ 1], A, K, m0, k0 + 32, tid);
      __builtin_prefetch(W + (size_t)(n0 + (lane & 15)) * K + k0 + 32, 0, 3);
    }
    v16bf a = frag_lds(sA[buf], lane);
#pragma unroll
    for (int j = 0; j < 4; ++j) {
      v16bf b = frag_glb(W, K, n0 + j * 16, k0, lane);
      acc[j] = WMMA_BF16(a, b, acc[j]);
    }
    wait_stage();
  }

#pragma unroll
  for (int j = 0; j < 4; ++j) {
    const int   nc = n0 + j * 16 + (lane & 15);
    const float bv = bias[nc];
#pragma unroll
    for (int r = 0; r < 8; ++r) {
      int row = m0 + ((lane >> 4) << 3) + r;
      C[(size_t)row * ldc + nc] = (__bf16)elu_f(acc[j][r] + bv);
    }
  }
}

// ---------------- fused GRU step ----------------
// Wave accumulates 6 tiles: gi_{r,z,n} = x@Wih^T at cols {n, D+n, 2D+n} and
// gh_{r,z,n} = h@Whh^T, then gates and writes h_new (+ feat slice of d_out).
// grid: (8, 16); block 256.
__global__ void gru_fused_kernel(const __bf16* __restrict__ X,
                                 const __bf16* __restrict__ Hb,
                                 const float*  __restrict__ Hf,
                                 const __bf16* __restrict__ Wih,
                                 const float*  __restrict__ bih,
                                 const __bf16* __restrict__ Whh,
                                 const float*  __restrict__ bhh,
                                 float*  __restrict__ HnF,
                                 __bf16* __restrict__ HnB,
                                 float*  __restrict__ feat,
                                 int t) {
  __shared__ __align__(16) __bf16 sX[2][16 * 32];
  __shared__ __align__(16) __bf16 sH[2][16 * 32];
  const int tid  = threadIdx.x;
  const int lane = tid & 31;
  const int wave = tid >> 5;
  const int m0   = blockIdx.y * 16;
  const int nw   = (blockIdx.x * 8 + wave) * 16;

  v8f accI[3] = {};
  v8f accH[3] = {};
  stage_tile(sX[0], X,  Dx, m0, 0, tid);
  stage_tile(sH[0], Hb, Dx, m0, 0, tid);
  wait_stage();

  const int nIter = Dx >> 5;
  for (int i = 0; i < nIter; ++i) {
    const int k0  = i << 5;
    const int buf = i & 1;
    if (i + 1 < nIter) {
      stage_tile(sX[buf ^ 1], X,  Dx, m0, k0 + 32, tid);
      stage_tile(sH[buf ^ 1], Hb, Dx, m0, k0 + 32, tid);
    }
    v16bf fx = frag_lds(sX[buf], lane);
    v16bf fh = frag_lds(sH[buf], lane);
#pragma unroll
    for (int g = 0; g < 3; ++g) {
      v16bf bi = frag_glb(Wih, Dx, nw + g * Dx, k0, lane);
      accI[g] = WMMA_BF16(fx, bi, accI[g]);
      v16bf bh = frag_glb(Whh, Dx, nw + g * Dx, k0, lane);
      accH[g] = WMMA_BF16(fh, bh, accH[g]);
    }
    wait_stage();
  }

  const int   nc  = nw + (lane & 15);
  const float bir = bih[nc], biz = bih[Dx + nc], bin = bih[2 * Dx + nc];
  const float bhr = bhh[nc], bhz = bhh[Dx + nc], bhn = bhh[2 * Dx + nc];
#pragma unroll
  for (int r = 0; r < 8; ++r) {
    int   row  = m0 + ((lane >> 4) << 3) + r;
    float rg   = sigmoid_f(accI[0][r] + bir + accH[0][r] + bhr);
    float zg   = sigmoid_f(accI[1][r] + biz + accH[1][r] + bhz);
    float ng   = tanhf(accI[2][r] + bin + rg * (accH[2][r] + bhn));
    float hOld = Hf[(size_t)row * Dx + nc];
    float hNew = (1.0f - zg) * ng + zg * hOld;
    HnF[(size_t)row * Dx + nc] = hNew;
    HnB[(size_t)row * Dx + nc] = (__bf16)hNew;
    feat[(size_t)row * (Tx * FEATD) + (size_t)t * FEATD + Sx + nc] = hNew;
  }
}

// ---------------- prior head: [256x1024] @ [64,1024]^T -> pm, softplus(ps)+0.1 ----------------
// grid: (1, 16), block 128 (4 waves cover N=64).
__global__ void prior_head_kernel(const __bf16* __restrict__ A,
                                  const __bf16* __restrict__ W,
                                  const float*  __restrict__ bias,
                                  float* __restrict__ pm, float* __restrict__ ps,
                                  int t) {
  __shared__ __align__(16) __bf16 sA[2][16 * 32];
  const int tid  = threadIdx.x;
  const int lane = tid & 31;
  const int wave = tid >> 5;
  const int m0   = blockIdx.y * 16;
  const int n0   = wave * 16;

  v8f acc = {};
  stage_tile(sA[0], A, Hx, m0, 0, tid);
  wait_stage();
  const int nIter = Hx >> 5;
  for (int i = 0; i < nIter; ++i) {
    const int k0  = i << 5;
    const int buf = i & 1;
    if (i + 1 < nIter) stage_tile(sA[buf ^ 1], A, Hx, m0, k0 + 32, tid);
    v16bf a = frag_lds(sA[buf], lane);
    v16bf b = frag_glb(W, Hx, n0, k0, lane);
    acc = WMMA_BF16(a, b, acc);
    wait_stage();
  }

  const int   nc = n0 + (lane & 15);
  const float bv = bias[nc];
#pragma unroll
  for (int r = 0; r < 8; ++r) {
    int    row = m0 + ((lane >> 4) << 3) + r;
    float  v   = acc[r] + bv;
    size_t o   = (size_t)row * (Tx * Sx) + (size_t)t * Sx;
    if (nc < Sx) pm[o + nc] = v;
    else         ps[o + nc - Sx] = softplus_f(v) + 0.1f;
  }
}

// ---------------- posterior head: qm/qs + reparameterized sample + feat/stoch ----------------
__global__ void post_head_kernel(const __bf16* __restrict__ A,
                                 const __bf16* __restrict__ W,
                                 const float*  __restrict__ bias,
                                 const float*  __restrict__ noise,
                                 float* __restrict__ qm, float* __restrict__ qs,
                                 float* __restrict__ stochF,
                                 float* __restrict__ feat,
                                 int t) {
  __shared__ __align__(16) __bf16 sA[2][16 * 32];
  __shared__ float sOut[16 * 64];
  const int tid  = threadIdx.x;
  const int lane = tid & 31;
  const int wave = tid >> 5;
  const int m0   = blockIdx.y * 16;
  const int n0   = wave * 16;

  v8f acc = {};
  stage_tile(sA[0], A, Hx, m0, 0, tid);
  wait_stage();
  const int nIter = Hx >> 5;
  for (int i = 0; i < nIter; ++i) {
    const int k0  = i << 5;
    const int buf = i & 1;
    if (i + 1 < nIter) stage_tile(sA[buf ^ 1], A, Hx, m0, k0 + 32, tid);
    v16bf a = frag_lds(sA[buf], lane);
    v16bf b = frag_glb(W, Hx, n0, k0, lane);
    acc = WMMA_BF16(a, b, acc);
    wait_stage();
  }

  const int   nc = n0 + (lane & 15);
  const float bv = bias[nc];
#pragma unroll
  for (int r = 0; r < 8; ++r) {
    int rl = ((lane >> 4) << 3) + r;
    sOut[rl * 64 + nc] = acc[r] + bv;
  }
  __syncthreads();

  for (int i = tid; i < 16 * Sx; i += blockDim.x) {
    int    rl  = i >> 5;
    int    s   = i & 31;
    int    row = m0 + rl;
    float  m   = sOut[rl * 64 + s];
    float  sd  = softplus_f(sOut[rl * 64 + Sx + s]) + 0.1f;
    float  eps = noise[(size_t)row * (Tx * Sx) + (size_t)t * Sx + s];
    float  st  = m + sd * eps;
    size_t o   = (size_t)row * (Tx * Sx) + (size_t)t * Sx + s;
    qm[o] = m;
    qs[o] = sd;
    stochF[row * Sx + s] = st;
    feat[(size_t)row * (Tx * FEATD) + (size_t)t * FEATD + s] = st;
  }
}

// ---------------- small elementwise kernels ----------------
__global__ void build_xin_kernel(const float* __restrict__ actions,
                                 const float* __restrict__ stochF,
                                 __bf16* __restrict__ xin, int t) {
  int idx = blockIdx.x * blockDim.x + threadIdx.x;  // Bx * 64
  int b = idx >> 6, c = idx & 63;
  float v = (c < Ax) ? actions[(size_t)b * (Tx * Ax) + (size_t)t * Ax + c]
                     : stochF[b * Sx + (c - Ax)];
  xin[idx] = (__bf16)v;
}

__global__ void build_cat_kernel(const __bf16* __restrict__ detB,
                                 const float* __restrict__ obs,
                                 __bf16* __restrict__ catb, int t) {
  int idx = blockIdx.x * blockDim.x + threadIdx.x;  // Bx * (Dx+Lx)
  int b = idx / (Dx + Lx), c = idx % (Dx + Lx);
  __bf16 v = (c < Dx) ? detB[b * Dx + c]
                      : (__bf16)obs[(size_t)b * (Tx * Lx) + (size_t)t * Lx + (c - Dx)];
  catb[idx] = v;
}

__global__ void cvt_bf16_kernel(const float* __restrict__ src, __bf16* __restrict__ dst, int n) {
  int i = blockIdx.x * blockDim.x + threadIdx.x;
  if (i < n) dst[i] = (__bf16)src[i];
}

// ---------------- host launcher ----------------
extern "C" void kernel_launch(void* const* d_in, const int* in_sizes, int n_in,
                              void* d_out, int out_size, void* d_ws, size_t ws_size,
                              hipStream_t stream) {
  (void)in_sizes; (void)n_in; (void)out_size; (void)ws_size;

  const float* obs = (const float*)d_in[0];
  const float* act = (const float*)d_in[1];
  const float* noi = (const float*)d_in[2];
  const float* w1  = (const float*)d_in[3];
  const float* b1  = (const float*)d_in[4];
  const float* wih = (const float*)d_in[5];
  const float* bih = (const float*)d_in[6];
  const float* whh = (const float*)d_in[7];
  const float* bhh = (const float*)d_in[8];
  const float* w2  = (const float*)d_in[9];
  const float* b2  = (const float*)d_in[10];
  const float* w3  = (const float*)d_in[11];
  const float* b3  = (const float*)d_in[12];
  const float* p1  = (const float*)d_in[13];
  const float* pb1 = (const float*)d_in[14];
  const float* p2  = (const float*)d_in[15];
  const float* pb2 = (const float*)d_in[16];

  float* out = (float*)d_out;
  const size_t BTS = (size_t)Bx * Tx * Sx;
  float* pm = out;
  float* ps = out + BTS;
  float* qm = out + 2 * BTS;
  float* qs = out + 3 * BTS;
  float* ft = out + 4 * BTS;

  char* wp = (char*)d_ws;
  auto carve = [&](size_t bytes) -> void* {
    void* p = (void*)wp;
    wp += (bytes + 255) & ~(size_t)255;
    return p;
  };

  __bf16* W1b  = (__bf16*)carve((size_t)Dx * 64 * 2);
  __bf16* Wihb = (__bf16*)carve((size_t)3 * Dx * Dx * 2);
  __bf16* Whhb = (__bf16*)carve((size_t)3 * Dx * Dx * 2);
  __bf16* W2b  = (__bf16*)carve((size_t)Hx * Dx * 2);
  __bf16* W3b  = (__bf16*)carve((size_t)2 * Sx * Hx * 2);
  __bf16* P1b  = (__bf16*)carve((size_t)Hx * (Dx + Lx) * 2);
  __bf16* P2b  = (__bf16*)carve((size_t)2 * Sx * Hx * 2);
  __bf16* xin  = (__bf16*)carve((size_t)Bx * 64 * 2);
  __bf16* xb   = (__bf16*)carve((size_t)Bx * Dx * 2);
  float*  dAf  = (float*)carve((size_t)Bx * Dx * 4);
  float*  dBf  = (float*)carve((size_t)Bx * Dx * 4);
  __bf16* dAb  = (__bf16*)carve((size_t)Bx * Dx * 2);
  __bf16* dBb  = (__bf16*)carve((size_t)Bx * Dx * 2);
  __bf16* phb  = (__bf16*)carve((size_t)Bx * Hx * 2);
  __bf16* catb = (__bf16*)carve((size_t)Bx * (Dx + Lx) * 2);
  __bf16* qhb  = (__bf16*)carve((size_t)Bx * Hx * 2);
  float*  stF  = (float*)carve((size_t)Bx * Sx * 4);

  auto cvt = [&](const float* s, __bf16* d, size_t n) {
    cvt_bf16_kernel<<<dim3((unsigned)((n + 255) / 256)), dim3(256), 0, stream>>>(s, d, (int)n);
  };
  cvt(w1,  W1b,  (size_t)Dx * 64);
  cvt(wih, Wihb, (size_t)3 * Dx * Dx);
  cvt(whh, Whhb, (size_t)3 * Dx * Dx);
  cvt(w2,  W2b,  (size_t)Hx * Dx);
  cvt(w3,  W3b,  (size_t)2 * Sx * Hx);
  cvt(p1,  P1b,  (size_t)Hx * (Dx + Lx));
  cvt(p2,  P2b,  (size_t)2 * Sx * Hx);

  hipMemsetAsync(stF, 0, (size_t)Bx * Sx * 4, stream);   // stoch_0 = 0
  hipMemsetAsync(dAf, 0, (size_t)Bx * Dx * 4, stream);   // det_0 = 0 (f32)
  hipMemsetAsync(dAb, 0, (size_t)Bx * Dx * 2, stream);   // det_0 = 0 (bf16)

  for (int t = 0; t < Tx; ++t) {
    float*  hOldF = (t & 1) ? dBf : dAf;
    __bf16* hOldB = (t & 1) ? dBb : dAb;
    float*  hNewF = (t & 1) ? dAf : dBf;
    __bf16* hNewB = (t & 1) ? dAb : dBb;

    build_xin_kernel<<<dim3(Bx * 64 / 256), dim3(256), 0, stream>>>(act, stF, xin, t);
    // x = elu([act, stoch] @ W1^T + b1)   M=256 N=1024 K=64
    gemm_bias_elu_kernel<<<dim3(Dx / 512, Bx / 16), dim3(256), 0, stream>>>(
        xin, 64, W1b, b1, xb, Dx);
    // fused GRU: det_new
    gru_fused_kernel<<<dim3(Dx / 128, Bx / 16), dim3(256), 0, stream>>>(
        xb, hOldB, hOldF, Wihb, bih, Whhb, bhh, hNewF, hNewB, ft, t);
    // ph = elu(det_new @ W2^T + b2)       M=256 N=1024 K=1024
    gemm_bias_elu_kernel<<<dim3(Hx / 512, Bx / 16), dim3(256), 0, stream>>>(
        hNewB, Dx, W2b, b2, phb, Hx);
    // prior stats
    prior_head_kernel<<<dim3(1, Bx / 16), dim3(128), 0, stream>>>(phb, W3b, b3, pm, ps, t);
    // qh = elu([det_new, obs] @ P1^T + pb1)  M=256 N=1024 K=2560
    build_cat_kernel<<<dim3(Bx * (Dx + Lx) / 256), dim3(256), 0, stream>>>(hNewB, obs, catb, t);
    gemm_bias_elu_kernel<<<dim3(Hx / 512, Bx / 16), dim3(256), 0, stream>>>(
        catb, Dx + Lx, P1b, pb1, qhb, Hx);
    // posterior stats + sample + feat/stoch
    post_head_kernel<<<dim3(1, Bx / 16), dim3(128), 0, stream>>>(
        qhb, P2b, pb2, noi, qm, qs, stF, ft, t);
  }
}
